// RNet_78847009620111
// MI455X (gfx1250) — matrix operand order repaired and checked by reading, not compiled, not run on hardware
//
#include <hip/hip_runtime.h>

typedef unsigned short u16;
typedef unsigned int   u32;
typedef __attribute__((ext_vector_type(16))) __bf16 v16bf;
typedef __attribute__((ext_vector_type(8)))  float  v8f;
typedef __attribute__((ext_vector_type(4)))  unsigned int u32x4;
typedef __attribute__((ext_vector_type(2)))  unsigned int u32x2;
typedef __attribute__((ext_vector_type(8)))  int i32x8;
typedef __attribute__((ext_vector_type(4)))  int i32x4;

union FragU { u32x4 q[2]; v16bf v; };

__device__ __forceinline__ u16 f2bf(float f) {
  union { float f; u32 u; } x; x.f = f;
  u32 u = x.u;
  u32 lsb = (u >> 16) & 1u;
  u += 0x7fffu + lsb;                 // round-to-nearest-even
  return (u16)(u >> 16);
}
__device__ __forceinline__ float sigm(float x) { return 1.f / (1.f + __expf(-x)); }

// ---------------------------------------------------------------- TDM helper
// Issue a 2D tensor_load_to_lds: tile_d1 rows x tile_d0 bf16 elements, row stride
// stride_elems (bf16 units), into LDS at lds_byte (padded: +4 dwords per 16 dwords
// -> LDS row stride of 40 halfs to keep ds_load_b128 fragment reads conflict-free).
// D# layout per CDNA5 ISA.
__device__ __forceinline__ void tdm_load_tile(unsigned lds_byte, const u16* gptr,
                                              int tile_d0, int tile_d1,
                                              long stride_elems) {
  unsigned long long ga = (unsigned long long)(uintptr_t)gptr;
  u32x4 g0 = {
    1u,                                           // count=1, user descriptor
    lds_byte,                                     // lds_addr
    (unsigned)(ga & 0xffffffffu),                 // global_addr[31:0]
    (unsigned)((ga >> 32) & 0x1ffffffu) | (2u << 30)  // global_addr[56:32], type=2
  };
  unsigned long long st = (unsigned long long)stride_elems;
  i32x8 g1 = {
    // data_size=1 (2B), pad_enable, pad_interval=3 (16 dwords), pad_amount=3 (4 dwords)
    (int)((1u << 16) | (1u << 20) | (3u << 22) | (3u << 25)),
    (int)0xffff0000u,                             // tensor_dim0 lo16 (dim0 = 0x7fffffff)
    (int)0xffff7fffu,                             // dim0 hi16 | tensor_dim1 lo16 (dim1 huge)
    (int)(0x7fffu | ((unsigned)tile_d0 << 16)),   // dim1 hi16 | tile_dim0
    (int)((unsigned)tile_d1 & 0xffffu),           // tile_dim1 | tile_dim2=0
    (int)(st & 0xffffffffu),                      // tensor_dim0_stride[31:0]
    (int)((st >> 32) & 0xffffu),                  // stride[47:32] | dim1_stride lo16 = 0
    0
  };
  i32x4 z4 = {};
#if __clang_major__ >= 23
  i32x8 z8 = {};
  __builtin_amdgcn_tensor_load_to_lds(g0, g1, z4, z4, z8, 0);
#else
  __builtin_amdgcn_tensor_load_to_lds(g0, g1, z4, z4, 0);
#endif
}

// ---------------------------------------------------------------- utility
__global__ void k_zero(u32* p, long n) {
  long i = (long)blockIdx.x * 256 + threadIdx.x;
  if (i < n) p[i] = 0u;
}

// f32 (N x K) -> bf16 (N x Kp), zero padded
__global__ void k_convert_w(const float* __restrict__ src, u16* __restrict__ dst,
                            int K, int Kp, long total) {
  long i = (long)blockIdx.x * 256 + threadIdx.x;
  if (i >= total) return;
  long n = i / Kp; int k = (int)(i - n * Kp);
  dst[i] = (k < K) ? f2bf(src[n * K + k]) : (u16)0;
}

// gather embedding rows -> bf16, padded to Kp (=320)
__global__ void k_gather(const int* __restrict__ idx, const float* __restrict__ emb,
                         u16* __restrict__ dst, int Kp, long total) {
  long i = (long)blockIdx.x * 256 + threadIdx.x;
  if (i >= total) return;
  long r = i / Kp; int k = (int)(i - r * Kp);
  int tok = idx[r];
  dst[i] = (k < 300) ? f2bf(emb[(size_t)tok * 300 + k]) : (u16)0;
}

// strided f32 -> bf16
__global__ void k_cvt_bf(const float* __restrict__ src, long sld,
                         u16* __restrict__ dst, long dld, long total, int cols) {
  long i = (long)blockIdx.x * 256 + threadIdx.x;
  if (i >= total) return;
  long r = i / cols; int c = (int)(i - r * cols);
  dst[r * dld + c] = f2bf(src[r * sld + c]);
}

// strided bf16 copy
__global__ void k_copy_bf(const u16* __restrict__ src, long sld,
                          u16* __restrict__ dst, long dld, long total, int cols) {
  long i = (long)blockIdx.x * 256 + threadIdx.x;
  if (i >= total) return;
  long r = i / cols; int c = (int)(i - r * cols);
  dst[r * dld + c] = src[r * sld + c];
}

// ---------------------------------------------------------------- GEMM (WMMA bf16 + TDM)
// C[M,N] = epilogue(A[M,K] * W[N,K]^T + bias)
// Block tile 32x128, 8 waves (2x4), each wave computes two 16x16 WMMA tiles
// sharing one A fragment. K stepped by 32 with double-buffered LDS; tiles are
// DMA'd global->LDS by the Tensor Data Mover (wave 0 issues, TENSORcnt sync),
// with TDM pad_enable producing the 40-half padded LDS row stride.
// epil==1: v = sigmoid(v) * Aux[m*ldaux+n]  (gate-mul)
__global__ __launch_bounds__(256) void k_gemm(
    const u16* __restrict__ A, int lda, const u16* __restrict__ W, int ldw,
    const float* __restrict__ bias, float* __restrict__ C, long ldc,
    u16* __restrict__ Obf, long ldbf, const float* __restrict__ Aux, int ldaux,
    int K, int epil)
{
  __shared__ __align__(16) u16 As[2][32 * 40];
  __shared__ __align__(16) u16 Ws[2][128 * 40];
  const int tid  = threadIdx.x;
  const int lane = tid & 31;
  const int wave = tid >> 5;
  const int wm = wave >> 2, wn = wave & 3;   // 2 x 4 wave grid
  const int half = lane >> 4, l16 = lane & 15;
  const long m0 = (long)blockIdx.y * 32;
  const long n0 = (long)blockIdx.x * 128;

  const int stages = K / 32;
  v8f acc0 = {}, acc1 = {};

  if (wave == 0) {
    tdm_load_tile((unsigned)(uintptr_t)&As[0][0], A + (size_t)m0 * lda, 32, 32, lda);
    tdm_load_tile((unsigned)(uintptr_t)&Ws[0][0], W + (size_t)n0 * ldw, 32, 128, ldw);
  }

  for (int s = 0; s < stages; ++s) {
    const int cb = s & 1;
    if (wave == 0) {
      if (s + 1 < stages) {
        const int k1 = (s + 1) * 32;
        tdm_load_tile((unsigned)(uintptr_t)&As[cb ^ 1][0], A + (size_t)m0 * lda + k1, 32, 32, lda);
        tdm_load_tile((unsigned)(uintptr_t)&Ws[cb ^ 1][0], W + (size_t)n0 * ldw + k1, 32, 128, ldw);
        __builtin_amdgcn_s_wait_tensorcnt(2);   // stage s landed; s+1 in flight
      } else {
        __builtin_amdgcn_s_wait_tensorcnt(0);
      }
    }
    __syncthreads();

    FragU a, b0, b1;
    const u16* arow = &As[cb][(wm * 16 + l16) * 40];
    a.q[0] = *(const u32x4*)(arow + 8 * half);        // elems 0..7 : K = 8h .. 8h+7
    a.q[1] = *(const u32x4*)(arow + 16 + 8 * half);   // elems 8..15: K = 16+8h ..
    const u16* brow0 = &Ws[cb][(wn * 32 + l16) * 40];
    b0.q[0] = *(const u32x4*)(brow0 + 8 * half);
    b0.q[1] = *(const u32x4*)(brow0 + 16 + 8 * half);
    const u16* brow1 = &Ws[cb][(wn * 32 + 16 + l16) * 40];
    b1.q[0] = *(const u32x4*)(brow1 + 8 * half);
    b1.q[1] = *(const u32x4*)(brow1 + 16 + 8 * half);
    acc0 = __builtin_amdgcn_wmma_f32_16x16x32_bf16(false, a.v, false, b0.v,
                                                   (short)0, acc0, false, false);
    acc1 = __builtin_amdgcn_wmma_f32_16x16x32_bf16(false, a.v, false, b1.v,
                                                   (short)0, acc1, false, false);
    __syncthreads();   // buffer cb may be overwritten next iteration
  }

#pragma unroll
  for (int tt = 0; tt < 2; ++tt) {
    const v8f acc = tt ? acc1 : acc0;
    const long n = n0 + wn * 32 + tt * 16 + l16;
    const float bv = bias ? bias[n] : 0.f;
#pragma unroll
    for (int j = 0; j < 8; ++j) {
      const long m = m0 + wm * 16 + j + 8 * half;
      float v = acc[j] + bv;
      if (epil == 1) v = sigm(v) * Aux[(size_t)m * ldaux + n];
      C[(size_t)m * ldc + n] = v;
      if (Obf) Obf[(size_t)m * ldbf + n] = f2bf(v);
    }
  }
}

// ---------------------------------------------------------------- attention score
// s[b*L+l] = ws . tanh(Hq[b*L+l] + u1[b] + u2[b]) + wsb ; one wave per row
__global__ __launch_bounds__(256) void k_score(
    const float* __restrict__ Hq, const float* __restrict__ u1, long ldu1,
    const float* __restrict__ u2, long ldu2,
    const float* __restrict__ wsw, const float* __restrict__ wsb,
    float* __restrict__ s, int L, int D)
{
  const int row  = blockIdx.x * 8 + (threadIdx.x >> 5);
  const int lane = threadIdx.x & 31;
  const int b = row / L;
  const float* hq = Hq + (size_t)row * D;
  const float* a1 = u1 + (size_t)b * ldu1;
  const float* a2 = u2 ? (u2 + (size_t)b * ldu2) : nullptr;
  float sum = 0.f;
  for (int d = lane; d < D; d += 32) {
    float v = hq[d] + a1[d] + (a2 ? a2[d] : 0.f);
    sum += tanhf(v) * wsw[d];
  }
  for (int o = 16; o; o >>= 1) sum += __shfl_xor(sum, o, 32);
  if (lane == 0) s[row] = sum + wsb[0];
}

// ---------------------------------------------------------------- softmax + context
// one block per batch element; c[b,:] = softmax(s[b,:]) @ H[b,:,:]
__global__ __launch_bounds__(256) void k_softmax_ctx(
    const float* __restrict__ s, const float* __restrict__ H,
    float* __restrict__ c, u16* __restrict__ cbf, float* __restrict__ probs,
    int L, int D)
{
  const int b = blockIdx.x;
  __shared__ float p[304];
  __shared__ float red[256];
  const float* sb = s + (size_t)b * L;
  float mx = -1e30f;
  for (int i = threadIdx.x; i < L; i += 256) mx = fmaxf(mx, sb[i]);
  red[threadIdx.x] = mx; __syncthreads();
  for (int st = 128; st; st >>= 1) {
    if (threadIdx.x < st) red[threadIdx.x] = fmaxf(red[threadIdx.x], red[threadIdx.x + st]);
    __syncthreads();
  }
  mx = red[0]; __syncthreads();
  float sm = 0.f;
  for (int i = threadIdx.x; i < L; i += 256) { float e = __expf(sb[i] - mx); p[i] = e; sm += e; }
  red[threadIdx.x] = sm; __syncthreads();
  for (int st = 128; st; st >>= 1) {
    if (threadIdx.x < st) red[threadIdx.x] += red[threadIdx.x + st];
    __syncthreads();
  }
  const float inv = 1.f / red[0];
  __syncthreads();
  for (int i = threadIdx.x; i < L; i += 256) {
    p[i] *= inv;
    if (probs) probs[(size_t)b * L + i] = p[i];
  }
  __syncthreads();
  for (int d = threadIdx.x; d < D; d += 256) {
    float acc = 0.f;
    const float* hb = H + (size_t)b * L * D + d;
    for (int l = 0; l < L; ++l) acc += p[l] * hb[(size_t)l * D];
    c[(size_t)b * D + d] = acc;
    if (cbf) cbf[(size_t)b * D + d] = f2bf(acc);
  }
}

// ---------------------------------------------------------------- GRU pointwise
__global__ void k_gru(const float* __restrict__ gi, long ldgi,
                      const float* __restrict__ gh, const float* __restrict__ hprev,
                      float* __restrict__ hout, u16* __restrict__ hbf,
                      float* __restrict__ yout, long ldy, int Hd)
{
  const int i = blockIdx.x * 256 + threadIdx.x;
  if (i >= 32 * Hd) return;
  const int b = i / Hd, j = i - b * Hd;
  const float* gib = gi + (size_t)b * ldgi;
  const float* ghb = gh + (size_t)b * 3 * Hd;
  const float r = sigm(gib[j] + ghb[j]);
  const float z = sigm(gib[Hd + j] + ghb[Hd + j]);
  const float n = tanhf(gib[2 * Hd + j] + r * ghb[2 * Hd + j]);
  const float h = (1.f - z) * n + z * hprev[(size_t)b * Hd + j];
  hout[(size_t)b * Hd + j] = h;
  hbf[(size_t)b * Hd + j] = f2bf(h);
  if (yout) yout[(size_t)b * ldy + j] = h;
}

// ha = mean(Q, axis=1) ; Q is (32,60,512)
__global__ void k_mean(const float* __restrict__ Q, float* __restrict__ ha, u16* __restrict__ habf) {
  const int i = blockIdx.x * 256 + threadIdx.x;
  if (i >= 32 * 512) return;
  const int b = i >> 9, d = i & 511;
  const float* q = Q + (size_t)b * 60 * 512 + d;
  float s = 0.f;
  for (int t = 0; t < 60; ++t) s += q[(size_t)t * 512];
  s *= (1.f / 60.f);
  ha[i] = s; habf[i] = f2bf(s);
}

// loss = ce(start,tag0)+ce(end,tag1); probs passed through log_softmax (as in reference)
__global__ void k_ce(const float* __restrict__ sp, const float* __restrict__ ep,
                     const int* __restrict__ tags, float* __restrict__ out) {
  __shared__ float vals[64];
  const int wave = threadIdx.x >> 5, lane = threadIdx.x & 31;
  for (int r = wave; r < 64; r += 8) {
    const int b = r & 31; const int isEnd = r >> 5;
    const float* p = (isEnd ? ep : sp) + (size_t)b * 300;
    const int t = tags[b * 2 + isEnd];
    float mx = -1e30f;
    for (int i = lane; i < 300; i += 32) mx = fmaxf(mx, p[i]);
    for (int o = 16; o; o >>= 1) mx = fmaxf(mx, __shfl_xor(mx, o, 32));
    float s = 0.f;
    for (int i = lane; i < 300; i += 32) s += __expf(p[i] - mx);
    for (int o = 16; o; o >>= 1) s += __shfl_xor(s, o, 32);
    if (lane == 0) vals[r] = -(p[t] - mx - __logf(s));
  }
  __syncthreads();
  if (threadIdx.x == 0) {
    float a = 0.f;
    for (int r = 0; r < 64; ++r) a += vals[r];
    out[0] = a / 32.f;
  }
}

// ================================================================ host
extern "C" void kernel_launch(void* const* d_in, const int* in_sizes, int n_in,
                              void* d_out, int out_size, void* d_ws, size_t ws_size,
                              hipStream_t stream)
{
  (void)in_sizes; (void)n_in; (void)out_size; (void)ws_size;
  const int B = 32, LQ = 60, LC = 300;
  const int NQ = B * LQ, NC = B * LC;

  char* cur = (char*)d_ws;
  auto alloc = [&](size_t bytes) -> void* {
    void* p = (void*)cur; cur += (bytes + 255) & ~(size_t)255; return p;
  };
  auto nb = [](long n) { return (unsigned)((n + 255) / 256); };

  // ---------------- weights -> bf16 (zero padded K) ----------------
  auto cvtW = [&](int idx, int N, int K, int Kp) -> u16* {
    u16* dst = (u16*)alloc((size_t)N * Kp * 2);
    long n = (long)N * Kp;
    k_convert_w<<<nb(n), 256, 0, stream>>>((const float*)d_in[idx], dst, K, Kp, n);
    return dst;
  };

  // insertion-order flatten of setup_inputs()
  u16 *e_wih[3][2], *e_whh[3][2];
  const float *e_bih[3][2], *e_bhh[3][2];
  for (int l = 0; l < 3; ++l)
    for (int d = 0; d < 2; ++d) {            // d=0: forward, d=1: backward
      const int base = 1 + l * 8 + d * 4;    // wih, whh, bih, bhh
      const int Kin = (l == 0) ? 300 : 512, Kp = (l == 0) ? 320 : 512;
      e_wih[l][d] = cvtW(base + 0, 768, Kin, Kp);
      e_whh[l][d] = cvtW(base + 1, 768, 256, 256);
      e_bih[l][d] = (const float*)d_in[base + 2];
      e_bhh[l][d] = (const float*)d_in[base + 3];
    }
  u16* g_wq = cvtW(25, 512, 512, 512);  const float* g_wq_b = (const float*)d_in[26];
  u16* g_wp = cvtW(27, 512, 512, 512);  const float* g_wp_b = (const float*)d_in[28];
  u16* g_wv = cvtW(29, 512, 512, 512);  const float* g_wv_b = (const float*)d_in[30];
  const float* g_ws_w = (const float*)d_in[31]; const float* g_ws_b = (const float*)d_in[32];
  u16* g_wg = cvtW(33, 512, 1024, 1024); const float* g_wg_b = (const float*)d_in[34];
  u16* g_wih = cvtW(35, 1536, 512, 512); u16* g_whh = cvtW(36, 1536, 512, 512);
  const float* g_bih = (const float*)d_in[37]; const float* g_bhh = (const float*)d_in[38];

  u16* s_wq = cvtW(39, 512, 512, 512);  const float* s_wq_b = (const float*)d_in[40];
  u16* s_wp = cvtW(41, 512, 512, 512);  const float* s_wp_b = (const float*)d_in[42];
  const float* s_ws_w = (const float*)d_in[45]; const float* s_ws_b = (const float*)d_in[46];
  u16* s_wg = cvtW(47, 512, 1024, 1024); const float* s_wg_b = (const float*)d_in[48];
  u16* sl_wih = cvtW(49, 1536, 512, 512); u16* sl_whh = cvtW(50, 1536, 512, 512);
  const float* sl_bih = (const float*)d_in[51]; const float* sl_bhh = (const float*)d_in[52];
  u16* sr_wih = cvtW(53, 1536, 512, 512); u16* sr_whh = cvtW(54, 1536, 512, 512);
  const float* sr_bih = (const float*)d_in[55]; const float* sr_bhh = (const float*)d_in[56];

  u16* p_wq = cvtW(57, 512, 1024, 1024); const float* p_wq_b = (const float*)d_in[58];
  u16* p_wp = cvtW(59, 512, 512, 512);   const float* p_wp_b = (const float*)d_in[60];
  const float* p_ws_w = (const float*)d_in[63]; const float* p_ws_b = (const float*)d_in[64];
  u16* p_wih = cvtW(65, 1536, 1024, 1024); u16* p_whh = cvtW(66, 1536, 512, 512);
  const float* p_bih = (const float*)d_in[67]; const float* p_bhh = (const float*)d_in[68];

  const float* emb    = (const float*)d_in[0];
  const int* question = (const int*)d_in[69];
  const int* context  = (const int*)d_in[70];
  const int* tags     = (const int*)d_in[71];

  // ---------------- activation buffers ----------------
  u16*   qembbf = (u16*)alloc((size_t)NQ * 320 * 2);
  u16*   cembbf = (u16*)alloc((size_t)NC * 320 * 2);
  float* r1     = (float*)alloc((size_t)NC * 1536 * 4);  // giAll / wpC / giAllRight
  float* Yq     = (float*)alloc((size_t)NQ * 512 * 4);   // Q
  float* Yc     = (float*)alloc((size_t)NC * 512 * 4);   // C (encoder out)
  u16*   Xbf    = (u16*)alloc((size_t)NC * 512 * 2);
  u16*   Qbf    = (u16*)alloc((size_t)NQ * 512 * 2);
  u16*   Cbf    = (u16*)alloc((size_t)NC * 512 * 2);
  float* HqQ    = (float*)alloc((size_t)NQ * 512 * 4);
  float* G      = (float*)alloc((size_t)NC * 512 * 4);   // gated layer out
  u16*   Gbf    = (u16*)alloc((size_t)NC * 512 * 2);
  float* HqS    = (float*)alloc((size_t)NC * 512 * 4);
  float* uAll   = (float*)alloc((size_t)NC * 512 * 4);
  float* cs     = (float*)alloc((size_t)NC * 512 * 4);
  u16*   csbf   = (u16*)alloc((size_t)NC * 512 * 2);
  float* C4     = (float*)alloc((size_t)NC * 1024 * 4);  // self-match out (l|r)
  u16*   C4bf   = (u16*)alloc((size_t)NC * 1024 * 2);
  float* HqP    = (float*)alloc((size_t)NC * 512 * 4);
  float* startp = (float*)alloc((size_t)32 * 300 * 4);
  float* endp   = (float*)alloc((size_t)32 * 300 * 4);
  float* hstate = (float*)alloc((size_t)32 * 512 * 4);
  u16*   hbf    = (u16*)alloc((size_t)32 * 512 * 2);
  float* ghbuf  = (float*)alloc((size_t)32 * 1536 * 4);
  float* gibuf  = (float*)alloc((size_t)32 * 1536 * 4);
  float* uvb    = (float*)alloc((size_t)32 * 512 * 4);
  float* sbuf   = (float*)alloc((size_t)32 * 300 * 4);
  float* cbuf   = (float*)alloc((size_t)32 * 1024 * 4);
  u16*   cbf1   = (u16*)alloc((size_t)32 * 1024 * 2);
  u16*   catbf  = (u16*)alloc((size_t)32 * 1024 * 2);
  float* gcf    = (float*)alloc((size_t)32 * 512 * 4);
  u16*   gcbf   = (u16*)alloc((size_t)32 * 512 * 2);

  auto gemm = [&](const u16* A, int lda, const u16* W, int ldw, const float* bias,
                  float* C, long ldc, u16* Obf, long ldbf, const float* Aux, int ldaux,
                  int M, int N, int K, int epil) {
    dim3 grid((unsigned)(N / 128), (unsigned)(M / 32));
    k_gemm<<<grid, 256, 0, stream>>>(A, lda, W, ldw, bias, C, ldc, Obf, ldbf, Aux, ldaux, K, epil);
  };
  auto zero = [&](void* p, size_t bytes) {
    k_zero<<<nb((long)(bytes / 4)), 256, 0, stream>>>((u32*)p, (long)(bytes / 4));
  };

  // ---------------- encoder ----------------
  auto run_dir = [&](const u16* X, int lda, int Kin, int T, const u16* wih, const u16* whh,
                     const float* bih, const float* bhh, float* Y, int off, bool rev) {
    gemm(X, lda, wih, Kin, bih, r1, 768, nullptr, 0, nullptr, 0, B * T, 768, Kin, 0);
    zero(hstate, 32 * 256 * 4); zero(hbf, 32 * 256 * 2);
    for (int s = 0; s < T; ++s) {
      const int t = rev ? (T - 1 - s) : s;
      gemm(hbf, 256, whh, 256, bhh, ghbuf, 768, nullptr, 0, nullptr, 0, 32, 768, 256, 0);
      k_gru<<<nb(32 * 256), 256, 0, stream>>>(r1 + (size_t)t * 768, (long)T * 768, ghbuf,
                                              hstate, hstate, hbf,
                                              Y + (size_t)t * 512 + off, (long)T * 512, 256);
    }
  };
  auto encode = [&](const u16* embbf, int T, float* Y) {
    const u16* X = embbf; int lda = 320, Kp = 320;
    for (int l = 0; l < 3; ++l) {
      run_dir(X, lda, Kp, T, e_wih[l][0], e_whh[l][0], e_bih[l][0], e_bhh[l][0], Y, 0, false);
      run_dir(X, lda, Kp, T, e_wih[l][1], e_whh[l][1], e_bih[l][1], e_bhh[l][1], Y, 256, true);
      if (l < 2) {
        k_cvt_bf<<<nb((long)B * T * 512), 256, 0, stream>>>(Y, 512, Xbf, 512, (long)B * T * 512, 512);
        X = Xbf; lda = 512; Kp = 512;
      }
    }
  };

  k_gather<<<nb((long)NQ * 320), 256, 0, stream>>>(question, emb, qembbf, 320, (long)NQ * 320);
  k_gather<<<nb((long)NC * 320), 256, 0, stream>>>(context, emb, cembbf, 320, (long)NC * 320);
  encode(qembbf, LQ, Yq);
  encode(cembbf, LC, Yc);

  // ---------------- gated attention layer ----------------
  k_cvt_bf<<<nb((long)NQ * 512), 256, 0, stream>>>(Yq, 512, Qbf, 512, (long)NQ * 512, 512);
  k_cvt_bf<<<nb((long)NC * 512), 256, 0, stream>>>(Yc, 512, Cbf, 512, (long)NC * 512, 512);
  gemm(Qbf, 512, g_wq, 512, g_wq_b, HqQ, 512, nullptr, 0, nullptr, 0, NQ, 512, 512, 0);
  float* wpC = r1;  // reuse big scratch
  gemm(Cbf, 512, g_wp, 512, g_wp_b, wpC, 512, nullptr, 0, nullptr, 0, NC, 512, 512, 0);
  zero(hstate, 32 * 512 * 4); zero(hbf, 32 * 512 * 2);   // vp = 0
  for (int t = 0; t < LC; ++t) {
    gemm(hbf, 512, g_wv, 512, g_wv_b, uvb, 512, nullptr, 0, nullptr, 0, 32, 512, 512, 0);
    k_score<<<NQ / 8, 256, 0, stream>>>(HqQ, wpC + (size_t)t * 512, (long)LC * 512,
                                        uvb, 512, g_ws_w, g_ws_b, sbuf, LQ, 512);
    k_softmax_ctx<<<32, 256, 0, stream>>>(sbuf, Yq, cbuf, nullptr, nullptr, LQ, 512);
    k_copy_bf<<<nb(32 * 512), 256, 0, stream>>>(Cbf + (size_t)t * 512, (long)LC * 512,
                                                catbf, 1024, (long)32 * 512, 512);
    k_cvt_bf<<<nb(32 * 512), 256, 0, stream>>>(cbuf, 512, catbf + 512, 1024, (long)32 * 512, 512);
    gemm(catbf, 1024, g_wg, 1024, g_wg_b, gcf, 512, gcbf, 512, cbuf, 512, 32, 512, 1024, 1);
    gemm(gcbf, 512, g_wih, 512, g_bih, gibuf, 1536, nullptr, 0, nullptr, 0, 32, 1536, 512, 0);
    gemm(hbf, 512, g_whh, 512, g_bhh, ghbuf, 1536, nullptr, 0, nullptr, 0, 32, 1536, 512, 0);
    k_gru<<<nb(32 * 512), 256, 0, stream>>>(gibuf, 1536, ghbuf, hstate, hstate, hbf,
                                            G + (size_t)t * 512, (long)LC * 512, 512);
  }

  // ---------------- self matching ----------------
  k_cvt_bf<<<nb((long)NC * 512), 256, 0, stream>>>(G, 512, Gbf, 512, (long)NC * 512, 512);
  gemm(Gbf, 512, s_wq, 512, s_wq_b, HqS, 512, nullptr, 0, nullptr, 0, NC, 512, 512, 0);
  gemm(Gbf, 512, s_wp, 512, s_wp_b, uAll, 512, nullptr, 0, nullptr, 0, NC, 512, 512, 0);
  zero(hstate, 32 * 512 * 4); zero(hbf, 32 * 512 * 2);
  for (int t = 0; t < LC; ++t) {
    k_score<<<NC / 8, 256, 0, stream>>>(HqS, uAll + (size_t)t * 512, (long)LC * 512,
                                        nullptr, 0, s_ws_w, s_ws_b, sbuf, LC, 512);
    k_softmax_ctx<<<32, 256, 0, stream>>>(sbuf, G, cbuf, nullptr, nullptr, LC, 512);
    k_copy_bf<<<nb(32 * 512), 256, 0, stream>>>(Gbf + (size_t)t * 512, (long)LC * 512,
                                                catbf, 1024, (long)32 * 512, 512);
    k_cvt_bf<<<nb(32 * 512), 256, 0, stream>>>(cbuf, 512, catbf + 512, 1024, (long)32 * 512, 512);
    gemm(catbf, 1024, s_wg, 1024, s_wg_b, cs + (size_t)t * 512, (long)LC * 512,
         csbf + (size_t)t * 512, (long)LC * 512, cbuf, 512, 32, 512, 1024, 1);
    gemm(csbf + (size_t)t * 512, LC * 512, sl_wih, 512, sl_bih, gibuf, 1536,
         nullptr, 0, nullptr, 0, 32, 1536, 512, 0);
    gemm(hbf, 512, sl_whh, 512, sl_bhh, ghbuf, 1536, nullptr, 0, nullptr, 0, 32, 1536, 512, 0);
    k_gru<<<nb(32 * 512), 256, 0, stream>>>(gibuf, 1536, ghbuf, hstate, hstate, hbf,
                                            C4 + (size_t)t * 1024, (long)LC * 1024, 512);
  }
  // right scan: all gi hoisted into one big GEMM
  gemm(csbf, 512, sr_wih, 512, sr_bih, r1, 1536, nullptr, 0, nullptr, 0, NC, 1536, 512, 0);
  zero(hstate, 32 * 512 * 4); zero(hbf, 32 * 512 * 2);
  for (int s = 0; s < LC; ++s) {
    const int t = LC - 1 - s;
    gemm(hbf, 512, sr_whh, 512, sr_bhh, ghbuf, 1536, nullptr, 0, nullptr, 0, 32, 1536, 512, 0);
    k_gru<<<nb(32 * 512), 256, 0, stream>>>(r1 + (size_t)t * 1536, (long)LC * 1536, ghbuf,
                                            hstate, hstate, hbf,
                                            C4 + (size_t)t * 1024 + 512, (long)LC * 1024, 512);
  }

  // ---------------- pointer network + loss ----------------
  k_cvt_bf<<<nb((long)NC * 1024), 256, 0, stream>>>(C4, 1024, C4bf, 1024, (long)NC * 1024, 1024);
  gemm(C4bf, 1024, p_wq, 1024, p_wq_b, HqP, 512, nullptr, 0, nullptr, 0, NC, 512, 1024, 0);
  k_mean<<<nb(32 * 512), 256, 0, stream>>>(Yq, hstate, hbf);   // ha
  gemm(hbf, 512, p_wp, 512, p_wp_b, uvb, 512, nullptr, 0, nullptr, 0, 32, 512, 512, 0);
  k_score<<<NC / 8, 256, 0, stream>>>(HqP, uvb, 512, nullptr, 0, p_ws_w, p_ws_b, sbuf, LC, 512);
  k_softmax_ctx<<<32, 256, 0, stream>>>(sbuf, C4, cbuf, cbf1, startp, LC, 1024);
  gemm(cbf1, 1024, p_wih, 1024, p_bih, gibuf, 1536, nullptr, 0, nullptr, 0, 32, 1536, 1024, 0);
  gemm(hbf, 512, p_whh, 512, p_bhh, ghbuf, 1536, nullptr, 0, nullptr, 0, 32, 1536, 512, 0);
  k_gru<<<nb(32 * 512), 256, 0, stream>>>(gibuf, 1536, ghbuf, hstate, hstate, hbf,
                                          nullptr, 0, 512);
  gemm(hbf, 512, p_wp, 512, p_wp_b, uvb, 512, nullptr, 0, nullptr, 0, 32, 512, 512, 0);
  k_score<<<NC / 8, 256, 0, stream>>>(HqP, uvb, 512, nullptr, 0, p_ws_w, p_ws_b, sbuf, LC, 512);
  k_softmax_ctx<<<32, 256, 0, stream>>>(sbuf, C4, cbuf, nullptr, endp, LC, 1024);

  k_ce<<<1, 256, 0, stream>>>(startp, endp, tags, (float*)d_out);
}
